// PhraseDecoderLayer_79731772883318
// MI455X (gfx1250) — matrix operand order repaired and checked by reading, not compiled, not run on hardware
//
#include <hip/hip_runtime.h>
#include <hip/hip_bf16.h>
#include <math.h>

#define BB 4
#define SS 1024
#define DD 1024
#define HH 16
#define HD 64
#define DFF 4096
#define MM 256

typedef __attribute__((ext_vector_type(16))) __bf16 v16bf;
typedef __attribute__((ext_vector_type(8)))  __bf16 v8bf;
typedef __attribute__((ext_vector_type(8)))  float  v8f;

static __device__ __forceinline__ unsigned short f2bf(float f) {
  unsigned int u = __builtin_bit_cast(unsigned int, f);
  u += 0x7FFFu + ((u >> 16) & 1u);
  return (unsigned short)(u >> 16);
}

static __device__ __forceinline__ v16bf join16(v8bf lo, v8bf hi) {
  v16bf r;
#pragma unroll
  for (int i = 0; i < 8; ++i) { r[i] = lo[i]; r[i + 8] = hi[i]; }
  return r;
}

// gfx1250 async DMA global -> LDS (16B per lane), tracked by ASYNCcnt.
static __device__ __forceinline__ void async_load_b128(unsigned lds_addr, const void* gaddr) {
  asm volatile("global_load_async_to_lds_b128 %0, %1, off"
               :: "v"(lds_addr), "v"(gaddr) : "memory");
}
static __device__ __forceinline__ void wait_async0() {
  asm volatile("s_wait_asynccnt 0x0" ::: "memory");
}

// ---------------------------------------------------------------------------
// elementwise helpers
// ---------------------------------------------------------------------------
__global__ void copy_f32_kernel(const float* __restrict__ src, float* __restrict__ dst, int n) {
  int i = blockIdx.x * blockDim.x + threadIdx.x;
  if (i < n) dst[i] = src[i];
}

__global__ void cvt_bf16_kernel(const float* __restrict__ src, unsigned short* __restrict__ dst, int n) {
  int i = blockIdx.x * blockDim.x + threadIdx.x;
  if (i < n) dst[i] = f2bf(src[i]);
}

// ---------------------------------------------------------------------------
// LayerNorm (fp32 in, bf16 out). One block per row of D=1024, 256 threads.
// ---------------------------------------------------------------------------
__global__ __launch_bounds__(256)
void ln_bf16_kernel(const float* __restrict__ x, const float* __restrict__ g,
                    const float* __restrict__ bta, unsigned short* __restrict__ y) {
  __shared__ float red[16];
  const int row = blockIdx.x, tid = threadIdx.x;
  const float* xr = x + (size_t)row * DD;
  float v[4];
  float s = 0.f;
#pragma unroll
  for (int i = 0; i < 4; ++i) { v[i] = xr[tid + i * 256]; s += v[i]; }
#pragma unroll
  for (int o = 16; o >= 1; o >>= 1) s += __shfl_xor(s, o, 32);
  if ((tid & 31) == 0) red[tid >> 5] = s;
  __syncthreads();
  if (tid == 0) {
    float t = 0.f;
#pragma unroll
    for (int i = 0; i < 8; ++i) t += red[i];
    red[8] = t / (float)DD;
  }
  __syncthreads();
  const float mean = red[8];
  float q = 0.f;
#pragma unroll
  for (int i = 0; i < 4; ++i) { float d = v[i] - mean; q += d * d; }
#pragma unroll
  for (int o = 16; o >= 1; o >>= 1) q += __shfl_xor(q, o, 32);
  if ((tid & 31) == 0) red[tid >> 5] = q;
  __syncthreads();
  if (tid == 0) {
    float t = 0.f;
#pragma unroll
    for (int i = 0; i < 8; ++i) t += red[i];
    red[9] = rsqrtf(t / (float)DD + 1e-5f);
  }
  __syncthreads();
  const float rstd = red[9];
  unsigned short* yr = y + (size_t)row * DD;
#pragma unroll
  for (int i = 0; i < 4; ++i) {
    int c = tid + i * 256;
    yr[c] = f2bf((v[i] - mean) * rstd * g[c] + bta[c]);
  }
}

// ---------------------------------------------------------------------------
// Tiled WMMA GEMM:  C[M,N] = A[M,K](bf16) * Bw[N,K](bf16)^T + bias
// mode 0: Cf = acc+bias   mode 1: Cf = resid+acc+bias   mode 2: Cbf = bf16(gelu(acc+bias))
// Block tile 128x128, K-step 64, double-buffered LDS filled by
// global_load_async_to_lds_b128; 8 waves (2Mx4N), wave tile 64x32.
// ---------------------------------------------------------------------------
#define GTILE 9216  // 128 * 72 elements per LDS buffer

__global__ __launch_bounds__(256)
void gemm_bf16_kernel(const unsigned short* __restrict__ A,
                      const unsigned short* __restrict__ Bw,
                      const float* __restrict__ bias,
                      const float* __restrict__ resid,
                      float* __restrict__ Cf,
                      unsigned short* __restrict__ Cbf,
                      int M, int N, int K, int mode) {
  __shared__ unsigned short As[2][GTILE];
  __shared__ unsigned short Bs[2][GTILE];
  const int tid = threadIdx.x;
  const int lane = tid & 31, wave = tid >> 5;
  const int wM = wave >> 2, wN = wave & 3;
  const int lh = lane >> 4, ln = lane & 15;
  const int bm = blockIdx.y * 128, bn = blockIdx.x * 128;
  (void)M;

  const unsigned asA = (unsigned)(size_t)(&As[0][0]);
  const unsigned asB = (unsigned)(size_t)(&Bs[0][0]);

  v8f acc[4][2];
#pragma unroll
  for (int m = 0; m < 4; ++m)
#pragma unroll
    for (int n = 0; n < 2; ++n)
#pragma unroll
      for (int i = 0; i < 8; ++i) acc[m][n][i] = 0.0f;

  auto load_tile = [&](int k0, int buf) {
#pragma unroll
    for (int i = 0; i < 4; ++i) {
      int c = tid + i * 256;                 // 1024 chunks of 8 bf16
      int r = c >> 3, kc = (c & 7) * 8;
      unsigned loff = (unsigned)(buf * GTILE + r * 72 + kc) * 2u;
      async_load_b128(asA + loff, A  + (size_t)(bm + r) * K + k0 + kc);
      async_load_b128(asB + loff, Bw + (size_t)(bn + r) * K + k0 + kc);
    }
  };

  const int ntile = K >> 6;
  load_tile(0, 0);
  wait_async0();
  __syncthreads();

  for (int kt = 0; kt < ntile; ++kt) {
    const int buf = kt & 1;
    if (kt + 1 < ntile) load_tile((kt + 1) << 6, buf ^ 1);  // overlap DMA with WMMA

#pragma unroll
    for (int f = 0; f < 2; ++f) {            // two 32-deep K sub-steps
      v16bf af[4], bfr[2];
#pragma unroll
      for (int m = 0; m < 4; ++m) {
        const unsigned short* p = &As[buf][(wM * 64 + m * 16 + ln) * 72 + f * 32];
        v8bf lo = *(const v8bf*)(p + lh * 8);
        v8bf hi = *(const v8bf*)(p + lh * 8 + 16);
        af[m] = join16(lo, hi);
      }
#pragma unroll
      for (int n = 0; n < 2; ++n) {
        const unsigned short* p = &Bs[buf][(wN * 32 + n * 16 + ln) * 72 + f * 32 + lh * 8];
        v8bf lo = *(const v8bf*)(p);
        v8bf hi = *(const v8bf*)(p + 16);
        bfr[n] = join16(lo, hi);
      }
#pragma unroll
      for (int m = 0; m < 4; ++m)
#pragma unroll
        for (int n = 0; n < 2; ++n)
          acc[m][n] = __builtin_amdgcn_wmma_f32_16x16x32_bf16(
              false, af[m], false, bfr[n], (short)0, acc[m][n], false, false);
    }
    wait_async0();
    __syncthreads();
  }

#pragma unroll
  for (int m = 0; m < 4; ++m)
#pragma unroll
    for (int n = 0; n < 2; ++n) {
      int col = bn + wN * 32 + n * 16 + ln;
      float bsv = bias ? bias[col] : 0.0f;
#pragma unroll
      for (int v = 0; v < 8; ++v) {
        int row = bm + wM * 64 + m * 16 + v + 8 * lh;
        float val = acc[m][n][v] + bsv;
        size_t idx = (size_t)row * N + col;
        if (mode == 0) {
          Cf[idx] = val;
        } else if (mode == 1) {
          Cf[idx] = resid[idx] + val;
        } else {
          float gg = 0.5f * val * (1.0f + erff(val * 0.70710678118654752f));
          Cbf[idx] = f2bf(gg);
        }
      }
    }
}

// ---------------------------------------------------------------------------
// RoPE + split QKV scratch (B*S, 3D) f32 -> Q/K bf16 (b,h,s,hd), V bf16
// TRANSPOSED (b,h,hd,s) so P.V B-fragments load contiguously.
// ---------------------------------------------------------------------------
__global__ void rope_split_kernel(const float* __restrict__ qkv,
                                  const float* __restrict__ cosb,
                                  const float* __restrict__ sinb,
                                  unsigned short* __restrict__ Qo,
                                  unsigned short* __restrict__ Ko,
                                  unsigned short* __restrict__ Vo) {
  int idx = blockIdx.x * blockDim.x + threadIdx.x;       // B*H*S*32
  if (idx >= BB * HH * SS * (HD / 2)) return;
  int i = idx & 31;
  int s = (idx >> 5) & (SS - 1);
  int h = (idx >> 15) & (HH - 1);
  int b = idx >> 19;
  size_t headQ = (size_t)b * HH + h;
  size_t base = ((size_t)b * SS + s) * (3 * DD) + h * HD + 2 * i;
  float c = cosb[s * 32 + i], sn = sinb[s * 32 + i];
  float q0 = qkv[base],          q1 = qkv[base + 1];
  float k0 = qkv[base + DD],     k1 = qkv[base + DD + 1];
  float v0 = qkv[base + 2 * DD], v1 = qkv[base + 2 * DD + 1];
  size_t ob = (headQ * SS + s) * HD + 2 * i;
  Qo[ob]     = f2bf(q0 * c - q1 * sn);
  Qo[ob + 1] = f2bf(q0 * sn + q1 * c);
  Ko[ob]     = f2bf(k0 * c - k1 * sn);
  Ko[ob + 1] = f2bf(k0 * sn + k1 * c);
  Vo[(headQ * HD + 2 * i) * SS + s]     = f2bf(v0);     // transposed
  Vo[(headQ * HD + 2 * i + 1) * SS + s] = f2bf(v1);
}

// cross-attn Q: scratch (B*S, D) f32 -> (b,h,s,hd) bf16
__global__ void reshape_q_kernel(const float* __restrict__ src, unsigned short* __restrict__ dst) {
  int idx = blockIdx.x * blockDim.x + threadIdx.x;       // B*S*D
  if (idx >= BB * SS * DD) return;
  int dcol = idx & (DD - 1);
  int s = (idx >> 10) & (SS - 1);
  int b = idx >> 20;
  int h = dcol >> 6, d = dcol & 63;
  dst[(((size_t)b * HH + h) * SS + s) * HD + d] = f2bf(src[(size_t)idx]);
}

// cross-attn K/V: scratch (B*MEM, 2D) f32 -> CK (b,h,m,hd), CV (b,h,hd,m) bf16
__global__ void reshape_kv_kernel(const float* __restrict__ src,
                                  unsigned short* __restrict__ CK,
                                  unsigned short* __restrict__ CV) {
  int idx = blockIdx.x * blockDim.x + threadIdx.x;       // B*MEM*2048
  if (idx >= BB * MM * 2 * DD) return;
  int col = idx & (2 * DD - 1);
  int m = (idx >> 11) & (MM - 1);
  int b = idx >> 19;
  float val = src[(size_t)idx];
  int which = col >> 10;
  int dc = col & (DD - 1);
  int h = dc >> 6, d = dc & 63;
  size_t headQ = (size_t)b * HH + h;
  if (which)
    CV[(headQ * HD + d) * MM + m] = f2bf(val);           // transposed
  else
    CK[(headQ * MM + m) * HD + d] = f2bf(val);
}

// ---------------------------------------------------------------------------
// Flash attention with WMMA.  Q/K in (b,h,s,hd) bf16, V TRANSPOSED (b,h,hd,skv).
// grid = (Sq/64, H, B), block = 128 (4 independent waves, 16 query rows each).
// No block barriers: the P tile is wave-private and same-wave DS ops are
// in-order.  O written bf16 into (b, q, h*HD+d) => (B*S, D).
// ---------------------------------------------------------------------------
__global__ __launch_bounds__(128)
void flash_attn_kernel(const unsigned short* __restrict__ Q,
                       const unsigned short* __restrict__ Km,
                       const unsigned short* __restrict__ Vm,
                       unsigned short* __restrict__ O,
                       int skv, int causal, float scale) {
  __shared__ unsigned short Pt[4 * 16 * 72];
  const int tid = threadIdx.x;
  const int wave = tid >> 5, lane = tid & 31;
  const int lh = lane >> 4, ln = lane & 15;
  const int qb = blockIdx.x, h = blockIdx.y, b = blockIdx.z;
  const size_t headQ = (size_t)b * HH + h;
  const int qrow0 = qb * 64 + wave * 16;
  const int khalf = lh * 8;

  // Q fragments (A-matrix, K-dim = hd split in two 32-chunks), kept in regs
  const unsigned short* Qp = Q + (headQ * SS + (qrow0 + ln)) * HD;
  v16bf qa[2];
#pragma unroll
  for (int f = 0; f < 2; ++f) {
    v8bf lo = *(const v8bf*)(Qp + f * 32 + khalf);
    v8bf hi = *(const v8bf*)(Qp + f * 32 + khalf + 16);
    qa[f] = join16(lo, hi);
  }

  v8f oacc[4];
#pragma unroll
  for (int n = 0; n < 4; ++n)
#pragma unroll
    for (int i = 0; i < 8; ++i) oacc[n][i] = 0.0f;
  float mrow[8], lrow[8];
#pragma unroll
  for (int v = 0; v < 8; ++v) { mrow[v] = -1e30f; lrow[v] = 0.0f; }

  const int nkb = causal ? (qb + 1) : (skv / 64);
  unsigned short* Pw = Pt + wave * 16 * 72;

  for (int j = 0; j < nkb; ++j) {
    const int kb0 = j * 64;
    // ---- S = (Q K^T) * scale ----
    v8f sacc[4];
#pragma unroll
    for (int n = 0; n < 4; ++n)
#pragma unroll
      for (int i = 0; i < 8; ++i) sacc[n][i] = 0.0f;
#pragma unroll
    for (int n = 0; n < 4; ++n) {
      const unsigned short* Kp = Km + (headQ * skv + (kb0 + n * 16 + ln)) * HD;
#pragma unroll
      for (int f = 0; f < 2; ++f) {
        v8bf lo = *(const v8bf*)(Kp + f * 32 + lh * 8);
        v8bf hi = *(const v8bf*)(Kp + f * 32 + lh * 8 + 16);
        v16bf kb = join16(lo, hi);
        sacc[n] = __builtin_amdgcn_wmma_f32_16x16x32_bf16(
            false, qa[f], false, kb, (short)0, sacc[n], false, false);
      }
    }

    // ---- online softmax in C-fragment layout ----
    float pv[4][8];
#pragma unroll
    for (int v = 0; v < 8; ++v) {
      const int row = qrow0 + v + 8 * lh;
      float mx = -1e30f;
#pragma unroll
      for (int n = 0; n < 4; ++n) {
        float s = sacc[n][v] * scale;
        const int col = kb0 + n * 16 + ln;
        if (causal && col > row) s = -1e30f;
        pv[n][v] = s;
        mx = fmaxf(mx, s);
      }
#pragma unroll
      for (int o = 8; o >= 1; o >>= 1) mx = fmaxf(mx, __shfl_xor(mx, o, 32));
      const float mnew = fmaxf(mrow[v], mx);
      const float alpha = __expf(mrow[v] - mnew);
      mrow[v] = mnew;
      float rs = 0.0f;
#pragma unroll
      for (int n = 0; n < 4; ++n) {
        float p = __expf(pv[n][v] - mnew);
        pv[n][v] = p;
        rs += p;
      }
#pragma unroll
      for (int o = 8; o >= 1; o >>= 1) rs += __shfl_xor(rs, o, 32);
      lrow[v] = lrow[v] * alpha + rs;
#pragma unroll
      for (int n = 0; n < 4; ++n) oacc[n][v] = oacc[n][v] * alpha;
    }

    // P (bf16) -> wave-private LDS tile (16x64, padded stride 72)
#pragma unroll
    for (int n = 0; n < 4; ++n)
#pragma unroll
      for (int v = 0; v < 8; ++v)
        Pw[(v + 8 * lh) * 72 + n * 16 + ln] = f2bf(pv[n][v]);

    // ---- O += P @ V  (V transposed: contiguous B-fragment loads) ----
#pragma unroll
    for (int f = 0; f < 2; ++f) {            // key-dim 32-chunks
      const unsigned short* pp = Pw + ln * 72 + f * 32;
      v8bf plo = *(const v8bf*)(pp + khalf);
      v8bf phi = *(const v8bf*)(pp + khalf + 16);
      v16bf pa = join16(plo, phi);
#pragma unroll
      for (int n = 0; n < 4; ++n) {
        const unsigned short* Vp =
            Vm + (headQ * HD + n * 16 + ln) * (size_t)skv + kb0 + f * 32 + lh * 8;
        v8bf lo = *(const v8bf*)(Vp);
        v8bf hi = *(const v8bf*)(Vp + 16);
        v16bf vb = join16(lo, hi);
        oacc[n] = __builtin_amdgcn_wmma_f32_16x16x32_bf16(
            false, pa, false, vb, (short)0, oacc[n], false, false);
      }
    }
  }

  // ---- normalize + write O (b, q, h*HD + d) ----
#pragma unroll
  for (int n = 0; n < 4; ++n)
#pragma unroll
    for (int v = 0; v < 8; ++v) {
      int row = qrow0 + v + 8 * lh;
      int col = n * 16 + ln;
      float o = oacc[n][v] / lrow[v];
      O[((size_t)b * SS + row) * DD + h * HD + col] = f2bf(o);
    }
}

// ---------------------------------------------------------------------------
// host-side launch sequence
// ---------------------------------------------------------------------------
extern "C" void kernel_launch(void* const* d_in, const int* in_sizes, int n_in,
                              void* d_out, int out_size, void* d_ws, size_t ws_size,
                              hipStream_t stream) {
  (void)in_sizes; (void)n_in; (void)out_size; (void)ws_size;
  const float* tgt      = (const float*)d_in[0];
  const float* memory   = (const float*)d_in[1];
  /* d_in[2] tgt_mask: causal mask applied analytically */
  const float* rope_cos = (const float*)d_in[3];
  const float* rope_sin = (const float*)d_in[4];
  const float* qkv_w    = (const float*)d_in[5];
  const float* qkv_b    = (const float*)d_in[6];
  const float* out_w    = (const float*)d_in[7];
  const float* out_b    = (const float*)d_in[8];
  const float* ca_in_w  = (const float*)d_in[9];
  const float* ca_in_b  = (const float*)d_in[10];
  const float* ca_out_w = (const float*)d_in[11];
  const float* ca_out_b = (const float*)d_in[12];
  const float* ffn_w1   = (const float*)d_in[13];
  const float* ffn_b1   = (const float*)d_in[14];
  const float* ffn_w2   = (const float*)d_in[15];
  const float* ffn_b2   = (const float*)d_in[16];
  const float* ln1_g    = (const float*)d_in[17];
  const float* ln1_b    = (const float*)d_in[18];
  const float* ln2_g    = (const float*)d_in[19];
  const float* ln2_b    = (const float*)d_in[20];
  const float* ln3_g    = (const float*)d_in[21];
  const float* ln3_b    = (const float*)d_in[22];

  float* out = (float*)d_out;
  char* ws = (char*)d_ws;
  size_t off = 0;
  auto carve = [&](size_t bytes) -> void* {
    void* p = ws + off;
    off += (bytes + 255) & ~(size_t)255;
    return p;
  };

  unsigned short* Wqkv   = (unsigned short*)carve((size_t)3 * DD * DD * 2);
  unsigned short* Wout   = (unsigned short*)carve((size_t)DD * DD * 2);
  unsigned short* Wcain  = (unsigned short*)carve((size_t)3 * DD * DD * 2);
  unsigned short* Wcaout = (unsigned short*)carve((size_t)DD * DD * 2);
  unsigned short* W1     = (unsigned short*)carve((size_t)DFF * DD * 2);
  unsigned short* W2     = (unsigned short*)carve((size_t)DD * DFF * 2);
  unsigned short* Memb   = (unsigned short*)carve((size_t)BB * MM * DD * 2);
  unsigned short* Xln    = (unsigned short*)carve((size_t)BB * SS * DD * 2);
  float*          Scr    = (float*)carve((size_t)BB * SS * DFF * 4);
  unsigned short* Qb     = (unsigned short*)carve((size_t)BB * HH * SS * HD * 2);
  unsigned short* Kb     = (unsigned short*)carve((size_t)BB * HH * SS * HD * 2);
  unsigned short* Vb     = (unsigned short*)carve((size_t)BB * HH * SS * HD * 2);
  unsigned short* CK     = (unsigned short*)carve((size_t)BB * HH * MM * HD * 2);
  unsigned short* CV     = (unsigned short*)carve((size_t)BB * HH * MM * HD * 2);
  unsigned short* AO     = (unsigned short*)carve((size_t)BB * SS * DD * 2);
  unsigned short* HFF    = (unsigned short*)carve((size_t)BB * SS * DFF * 2);

  const int TOK = BB * SS;                     // 4096 token rows
  const float scale = 0.125f;                  // 1/sqrt(64)
  auto cdiv = [](int a, int b) { return (a + b - 1) / b; };

  // residual accumulator <- tgt
  copy_f32_kernel<<<cdiv(TOK * DD, 256), 256, 0, stream>>>(tgt, out, TOK * DD);

  // weight / memory conversion fp32 -> bf16
  cvt_bf16_kernel<<<cdiv(3 * DD * DD, 256), 256, 0, stream>>>(qkv_w,   Wqkv,   3 * DD * DD);
  cvt_bf16_kernel<<<cdiv(DD * DD, 256),     256, 0, stream>>>(out_w,   Wout,   DD * DD);
  cvt_bf16_kernel<<<cdiv(3 * DD * DD, 256), 256, 0, stream>>>(ca_in_w, Wcain,  3 * DD * DD);
  cvt_bf16_kernel<<<cdiv(DD * DD, 256),     256, 0, stream>>>(ca_out_w,Wcaout, DD * DD);
  cvt_bf16_kernel<<<cdiv(DFF * DD, 256),    256, 0, stream>>>(ffn_w1,  W1,     DFF * DD);
  cvt_bf16_kernel<<<cdiv(DD * DFF, 256),    256, 0, stream>>>(ffn_w2,  W2,     DD * DFF);
  cvt_bf16_kernel<<<cdiv(BB * MM * DD, 256),256, 0, stream>>>(memory,  Memb,   BB * MM * DD);

  // ---- self attention ----
  ln_bf16_kernel<<<TOK, 256, 0, stream>>>(out, ln1_g, ln1_b, Xln);
  gemm_bf16_kernel<<<dim3(3 * DD / 128, TOK / 128), 256, 0, stream>>>(
      Xln, Wqkv, qkv_b, nullptr, Scr, nullptr, TOK, 3 * DD, DD, 0);
  rope_split_kernel<<<cdiv(BB * HH * SS * (HD / 2), 256), 256, 0, stream>>>(
      Scr, rope_cos, rope_sin, Qb, Kb, Vb);
  flash_attn_kernel<<<dim3(SS / 64, HH, BB), 128, 0, stream>>>(
      Qb, Kb, Vb, AO, SS, 1, scale);
  gemm_bf16_kernel<<<dim3(DD / 128, TOK / 128), 256, 0, stream>>>(
      AO, Wout, out_b, out, out, nullptr, TOK, DD, DD, 1);

  // ---- cross attention ----
  ln_bf16_kernel<<<TOK, 256, 0, stream>>>(out, ln2_g, ln2_b, Xln);
  gemm_bf16_kernel<<<dim3(DD / 128, TOK / 128), 256, 0, stream>>>(
      Xln, Wcain, ca_in_b, nullptr, Scr, nullptr, TOK, DD, DD, 0);
  reshape_q_kernel<<<cdiv(BB * SS * DD, 256), 256, 0, stream>>>(Scr, Qb);
  gemm_bf16_kernel<<<dim3(2 * DD / 128, BB * MM / 128), 256, 0, stream>>>(
      Memb, Wcain + (size_t)DD * DD, ca_in_b + DD, nullptr, Scr, nullptr,
      BB * MM, 2 * DD, DD, 0);
  reshape_kv_kernel<<<cdiv(BB * MM * 2 * DD, 256), 256, 0, stream>>>(Scr, CK, CV);
  flash_attn_kernel<<<dim3(SS / 64, HH, BB), 128, 0, stream>>>(
      Qb, CK, CV, AO, MM, 0, scale);
  gemm_bf16_kernel<<<dim3(DD / 128, TOK / 128), 256, 0, stream>>>(
      AO, Wcaout, ca_out_b, out, out, nullptr, TOK, DD, DD, 1);

  // ---- FFN ----
  ln_bf16_kernel<<<TOK, 256, 0, stream>>>(out, ln3_g, ln3_b, Xln);
  gemm_bf16_kernel<<<dim3(DFF / 128, TOK / 128), 256, 0, stream>>>(
      Xln, W1, ffn_b1, nullptr, nullptr, HFF, TOK, DFF, DD, 2);
  gemm_bf16_kernel<<<dim3(DD / 128, TOK / 128), 256, 0, stream>>>(
      HFF, W2, ffn_b2, out, out, nullptr, TOK, DD, DFF, 1);
}
